// TSE_15118284882739
// MI455X (gfx1250) — matrix operand order repaired and checked by reading, not compiled, not run on hardware
//
#include <hip/hip_runtime.h>
#include <math.h>

#define BB 96        // b*t
#define TT 12
#define CH 64
#define NP 512
#define OC 64
#define KM 10
#define KC 8
#define EPSX 1e-5f

typedef __attribute__((ext_vector_type(16))) _Float16 v16h;
typedef __attribute__((ext_vector_type(8)))  float    v8f;

__device__ __forceinline__ v8f wmma32f16(v16h a, v16h b, v8f c) {
  // D = A(16x32 f16) * B(32x16 f16) + C(16x16 f32)
  return __builtin_amdgcn_wmma_f32_16x16x32_f16(false, a, false, b, (short)0, c,
                                                false, false);
}

// ---------------------------------------------------------------------------
// x0 = x + BN(conv_erase(mean_n(x)))   (erase with masks=soft=ones)
// ---------------------------------------------------------------------------
__global__ void k_erase0(const float* __restrict__ x, const float* __restrict__ w_e,
                         const float* __restrict__ cb, const float* __restrict__ ga,
                         const float* __restrict__ be, const float* __restrict__ mm,
                         const float* __restrict__ vv, float* __restrict__ x0) {
  const int b = blockIdx.x, t = threadIdx.x;   // 64 threads, t = channel
  __shared__ float res[CH];
  __shared__ float yv[CH];
  const float* xb = x + (size_t)b * CH * NP;
  float s = 0.f;
  for (int i = 0; i < NP; ++i) s += xb[t * NP + i];
  res[t] = s * (1.0f / NP);
  __syncthreads();
  float a = cb[t];
  for (int c = 0; c < CH; ++c) a += w_e[t * CH + c] * res[c];
  yv[t] = (a - mm[t]) * (ga[t] * rsqrtf(vv[t] + EPSX)) + be[t];
  __syncthreads();
  float* ob = x0 + (size_t)b * CH * NP;
  for (int c = 0; c < CH; ++c)
    for (int i = t; i < NP; i += 64)
      ob[c * NP + i] = xb[c * NP + i] + yv[c];
}

// ---------------------------------------------------------------------------
// x1seq[b,t] = x[b, min(t+1, T-1)]
// ---------------------------------------------------------------------------
__global__ void k_x1seq(const float* __restrict__ x, float* __restrict__ x1s) {
  size_t e = (size_t)blockIdx.x * blockDim.x + threadIdx.x;
  const size_t tot = (size_t)BB * CH * NP;
  if (e >= tot) return;
  int bb = (int)(e / (CH * NP));
  int rem = (int)(e % (CH * NP));
  int b = bb / TT, t = bb % TT;
  int ts = (t + 1 < TT) ? t + 1 : TT - 1;
  x1s[e] = x[((size_t)(b * TT + ts)) * CH * NP + rem];
}

// ---------------------------------------------------------------------------
// xx[b,i] = sum_c X[b,c,i]^2
// ---------------------------------------------------------------------------
__global__ void k_xx(const float* __restrict__ x, float* __restrict__ xx) {
  int e = blockIdx.x * blockDim.x + threadIdx.x;
  if (e >= BB * NP) return;
  int b = e / NP, i = e % NP;
  const float* xb = x + (size_t)b * CH * NP;
  float s = 0.f;
  for (int c = 0; c < CH; ++c) { float u = xb[c * NP + i]; s += u * u; }
  xx[e] = s;
}

// ---------------------------------------------------------------------------
// kNN: one wave per (batch, 16-row tile). Gram via WMMA f16 -> pd tile ->
// per-row top-k insertion in LDS.  pd = 2*G - xx[row] - xx[col].
// ---------------------------------------------------------------------------
__global__ void __launch_bounds__(32)
k_knn(const float* __restrict__ x, const float* __restrict__ xx,
      int* __restrict__ idx, int kk) {
  const int b = blockIdx.y, m0 = blockIdx.x * 16;
  const int lane = threadIdx.x & 31;
  const int hi = lane >> 4, lo = lane & 15;
  const float* xb = x + (size_t)b * CH * NP;
  const float* xxb = xx + (size_t)b * NP;
  __shared__ float tile[16 * 16];
  __shared__ float tv[16][KM];
  __shared__ int   ti[16][KM];
  if (lane < 16)
    for (int j = 0; j < kk; ++j) { tv[lane][j] = -3.4e38f; ti[lane][j] = 0; }
  // A fragment: rows = points m0..m0+15, K = channels (2 chunks of 32)
  v16h a[2];
#pragma unroll
  for (int ks = 0; ks < 2; ++ks)
#pragma unroll
    for (int h = 0; h < 16; ++h) {
      int koff = (h < 8) ? (hi * 8 + h) : (16 + hi * 8 + (h - 8));
      a[ks][h] = (_Float16)xb[(ks * 32 + koff) * NP + (m0 + lo)];
    }
  __syncthreads();
  for (int nt = 0; nt < NP / 16; ++nt) {
    const int n0 = nt * 16;
    v8f acc = {};
#pragma unroll
    for (int ks = 0; ks < 2; ++ks) {
      v16h bf;
#pragma unroll
      for (int e = 0; e < 16; ++e)
        bf[e] = (_Float16)xb[(ks * 32 + ((e < 8) ? (hi * 8 + e) : (16 + hi * 8 + (e - 8)))) * NP + (n0 + lo)];
      acc = wmma32f16(a[ks], bf, acc);
    }
#pragma unroll
    for (int r = 0; r < 8; ++r) {
      int M = r + 8 * hi;
      tile[M * 16 + lo] = 2.0f * acc[r] - xxb[m0 + M] - xxb[n0 + lo];
    }
    __syncthreads();
    if (lane < 16) {
#pragma unroll
      for (int j = 0; j < 16; ++j) {
        float val = tile[lane * 16 + j];
        if (val > tv[lane][kk - 1]) {
          int p = kk - 1;
          while (p > 0 && tv[lane][p - 1] < val) {  // strict: stable ties
            tv[lane][p] = tv[lane][p - 1]; ti[lane][p] = ti[lane][p - 1]; --p;
          }
          tv[lane][p] = val; ti[lane][p] = n0 + j;
        }
      }
    }
    __syncthreads();
  }
  if (lane < 16)
    for (int j = 0; j < kk; ++j)
      idx[((size_t)b * NP + m0 + lane) * kk + j] = ti[lane][j];
}

// ---------------------------------------------------------------------------
// Fused edge-conv branch: gather graph features to LDS (f16), WMMA GEMM
// (64x160 out of K=128), BN + LeakyReLU, max over k.  4 waves / block.
// ---------------------------------------------------------------------------
__global__ void __launch_bounds__(128)
k_branch(const float* __restrict__ x, const int* __restrict__ idx,
         const float* __restrict__ w, const float* __restrict__ ga,
         const float* __restrict__ cb, const float* __restrict__ mm,
         const float* __restrict__ vv, float* __restrict__ y) {
  const int b = blockIdx.y, p0 = blockIdx.x * 16;
  const int tid = threadIdx.x;
  const int lane = tid & 31, wv = tid >> 5;
  const int hi = lane >> 4, lo = lane & 15;
  __shared__ _Float16 s_w[OC * 128];       // 16 KB
  __shared__ _Float16 s_feat[128 * 160];   // 40 KB, re-aliased as f32 y-tile
  __shared__ int s_idx[16 * KM];
  const float* xb = x + (size_t)b * CH * NP;
  for (int i = tid; i < OC * 128; i += 128) s_w[i] = (_Float16)w[i];
  for (int i = tid; i < 16 * KM; i += 128)
    s_idx[i] = idx[((size_t)b * NP + p0) * KM + i];
  __syncthreads();
  // rows 0..63: neighbor - center, rows 64..127: center
  for (int e = tid; e < CH * 160; e += 128) {
    int ch = e / 160, col = e % 160;
    int p = col / KM, q = col % KM;
    float ct = xb[ch * NP + p0 + p];
    float nb = xb[ch * NP + s_idx[p * KM + q]];
    s_feat[ch * 160 + col]        = (_Float16)(nb - ct);
    s_feat[(CH + ch) * 160 + col] = (_Float16)ct;
  }
  __syncthreads();
  const int o0 = wv * 16;                  // wave's output-channel tile
  v16h a[4];
#pragma unroll
  for (int ks = 0; ks < 4; ++ks)
#pragma unroll
    for (int h = 0; h < 16; ++h) {
      int koff = (h < 8) ? (hi * 8 + h) : (16 + hi * 8 + (h - 8));
      a[ks][h] = s_w[(o0 + lo) * 128 + ks * 32 + koff];
    }
  v8f accs[10];
#pragma unroll
  for (int nt = 0; nt < 10; ++nt) {
    v8f acc = {};
#pragma unroll
    for (int ks = 0; ks < 4; ++ks) {
      v16h bf;
#pragma unroll
      for (int e = 0; e < 16; ++e)
        bf[e] = s_feat[(ks * 32 + ((e < 8) ? (hi * 8 + e) : (16 + hi * 8 + (e - 8)))) * 160 + nt * 16 + lo];
      acc = wmma32f16(a[ks], bf, acc);
    }
    accs[nt] = acc;
  }
  __syncthreads();                         // all waves done reading s_feat
  float* s_y = (float*)s_feat;             // 64*160 f32 == 40 KB, exact reuse
#pragma unroll
  for (int nt = 0; nt < 10; ++nt)
#pragma unroll
    for (int r = 0; r < 8; ++r) {
      int o = o0 + r + 8 * hi;
      float val = accs[nt][r];
      val = (val - mm[o]) * (ga[o] * rsqrtf(vv[o] + EPSX)) + cb[o];
      val = (val > 0.f) ? val : 0.2f * val;
      s_y[o * 160 + nt * 16 + lo] = val;
    }
  __syncthreads();
  for (int e = tid; e < OC * 16; e += 128) {
    int o = e >> 4, p = e & 15;
    float mx = -3.4e38f;
    for (int q = 0; q < KM; ++q) mx = fmaxf(mx, s_y[o * 160 + p * KM + q]);
    y[((size_t)b * OC + o) * NP + p0 + p] = mx;
  }
}

// ---------------------------------------------------------------------------
// p[b, o] = max_n y, p[b, OC+o] = mean_n y; optional xvec (max) output.
// add==0 overwrites d_out (p0), add==1 accumulates (p0+p1).
// ---------------------------------------------------------------------------
__global__ void k_reduce_y(const float* __restrict__ y, float* __restrict__ out,
                           float* __restrict__ xvec, int add) {
  const int b = blockIdx.x, o = threadIdx.x;   // 64 threads
  const float* yb = y + ((size_t)b * OC + o) * NP;
  float mx = -3.4e38f, s = 0.f;
  for (int i = 0; i < NP; ++i) { float v = yb[i]; mx = fmaxf(mx, v); s += v; }
  s *= (1.0f / NP);
  if (xvec) xvec[b * OC + o] = mx;
  if (add) { out[b * 2 * OC + o] += mx; out[b * 2 * OC + OC + o] += s; }
  else     { out[b * 2 * OC + o]  = mx; out[b * 2 * OC + OC + o]  = s; }
}

// ---------------------------------------------------------------------------
// corre_binar: q = w_reduce @ x0vec; f = scale * q.x2; fk = f + sum_k f[idx8];
// argmax -> masks (1 everywhere, 0 at argmax point + its 8 neighbors).
// ---------------------------------------------------------------------------
__global__ void k_corre(const float* __restrict__ xvec, const float* __restrict__ x2,
                        const float* __restrict__ wr, const int* __restrict__ idx8,
                        float* __restrict__ fbuf, float* __restrict__ mask) {
  const int b = blockIdx.x, t = threadIdx.x;   // 64 threads
  __shared__ float sq[CH];
  __shared__ float sf[NP];
  __shared__ float rv[64];
  __shared__ int   ri[64];
  const float* x2b = x2 + (size_t)b * CH * NP;
  float a = 0.f;
  for (int c = 0; c < CH; ++c) a += wr[t * CH + c] * xvec[b * CH + c];
  sq[t] = a;
  __syncthreads();
  const float scale = 0.125f;                  // 1/sqrt(64)
  for (int i = t; i < NP; i += 64) {
    float s = 0.f;
    for (int c = 0; c < CH; ++c) s += sq[c] * x2b[c * NP + i];
    s *= scale;
    sf[i] = s;
    fbuf[(size_t)b * NP + i] = s;
    mask[(size_t)b * NP + i] = 1.0f;
  }
  __syncthreads();
  float best = -3.4e38f; int bi = NP;
  const int* i8 = idx8 + (size_t)b * NP * KC;
  for (int i = t; i < NP; i += 64) {
    float fk = sf[i];
    for (int q = 0; q < KC; ++q) fk += sf[i8[i * KC + q]];
    if (fk > best || (fk == best && i < bi)) { best = fk; bi = i; }
  }
  rv[t] = best; ri[t] = bi;
  __syncthreads();
  if (t == 0) {
    for (int j = 1; j < 64; ++j)
      if (rv[j] > best || (rv[j] == best && ri[j] < bi)) { best = rv[j]; bi = ri[j]; }
    mask[(size_t)b * NP + bi] = 0.f;
    for (int q = 0; q < KC; ++q) mask[(size_t)b * NP + i8[bi * KC + q]] = 0.f;
  }
}

// ---------------------------------------------------------------------------
// x1e = x1s*mask + BN(conv_erase(attention-pooled x1s)) broadcast
// ---------------------------------------------------------------------------
__global__ void k_erase1(const float* __restrict__ x1s, const float* __restrict__ mask,
                         const float* __restrict__ fbuf, const float* __restrict__ w_e,
                         const float* __restrict__ cb, const float* __restrict__ ga,
                         const float* __restrict__ be, const float* __restrict__ mm,
                         const float* __restrict__ vv, float* __restrict__ x1e) {
  const int b = blockIdx.x, t = threadIdx.x;   // 64 threads
  __shared__ float sm[NP];
  __shared__ float red[64];
  __shared__ float res[CH], yv[CH];
  __shared__ float s_max, s_sum;
  const float* xb = x1s + (size_t)b * CH * NP;
  const float* mk = mask + (size_t)b * NP;
  const float* fb = fbuf + (size_t)b * NP;
  float mx = -3.4e38f;
  for (int i = t; i < NP; i += 64) {
    float s = fb[i] - (1.0f - mk[i]) * 1e8f;
    sm[i] = s;
    mx = fmaxf(mx, s);
  }
  red[t] = mx; __syncthreads();
  if (t == 0) { float m2 = red[0]; for (int j = 1; j < 64; ++j) m2 = fmaxf(m2, red[j]); s_max = m2; }
  __syncthreads();
  mx = s_max;
  float sum = 0.f;
  for (int i = t; i < NP; i += 64) { float e = expf(sm[i] - mx); sm[i] = e; sum += e; }
  red[t] = sum; __syncthreads();
  if (t == 0) { float a = 0.f; for (int j = 0; j < 64; ++j) a += red[j]; s_sum = a; }
  __syncthreads();
  const float inv = 1.0f / s_sum;
  float rs = 0.f;
  for (int i = 0; i < NP; ++i) rs += xb[t * NP + i] * sm[i];
  res[t] = rs * inv;
  __syncthreads();
  float a = cb[t];
  for (int c = 0; c < CH; ++c) a += w_e[t * CH + c] * res[c];
  yv[t] = (a - mm[t]) * (ga[t] * rsqrtf(vv[t] + EPSX)) + be[t];
  __syncthreads();
  float* ob = x1e + (size_t)b * CH * NP;
  for (int c = 0; c < CH; ++c)
    for (int i = t; i < NP; i += 64)
      ob[c * NP + i] = xb[c * NP + i] * mk[i] + yv[c];
}

// ---------------------------------------------------------------------------
extern "C" void kernel_launch(void* const* d_in, const int* in_sizes, int n_in,
                              void* d_out, int out_size, void* d_ws, size_t ws_size,
                              hipStream_t stream) {
  (void)in_sizes; (void)n_in; (void)out_size; (void)ws_size;
  const float* x        = (const float*)d_in[0];
  const float* w_reduce = (const float*)d_in[1];
  const float* w_erase  = (const float*)d_in[2];
  const float* b_erase  = (const float*)d_in[3];
  const float* g_erase  = (const float*)d_in[4];
  const float* be_erase = (const float*)d_in[5];
  const float* m_erase  = (const float*)d_in[6];
  const float* v_erase  = (const float*)d_in[7];
  const float* w0 = (const float*)d_in[8];
  const float* g0 = (const float*)d_in[9];
  const float* b0 = (const float*)d_in[10];
  const float* m0 = (const float*)d_in[11];
  const float* v0 = (const float*)d_in[12];
  const float* w1 = (const float*)d_in[13];
  const float* g1 = (const float*)d_in[14];
  const float* b1 = (const float*)d_in[15];
  const float* m1 = (const float*)d_in[16];
  const float* v1 = (const float*)d_in[17];
  float* out = (float*)d_out;

  char* ws = (char*)d_ws;
  size_t off = 0;
  auto alloc = [&](size_t bytes) -> void* {
    void* p = ws + off;
    off += (bytes + 255) & ~(size_t)255;
    return p;
  };
  const size_t feat_f = (size_t)BB * CH * NP;
  float* x0   = (float*)alloc(sizeof(float) * feat_f);
  float* x1s  = (float*)alloc(sizeof(float) * feat_f);
  float* x1e  = (float*)alloc(sizeof(float) * feat_f);
  float* ybuf = (float*)alloc(sizeof(float) * (size_t)BB * OC * NP);
  float* xxb  = (float*)alloc(sizeof(float) * (size_t)BB * NP);
  int*   idxm = (int*)alloc(sizeof(int) * (size_t)BB * NP * KM);
  int*   idx8 = (int*)alloc(sizeof(int) * (size_t)BB * NP * KC);
  float* xvec = (float*)alloc(sizeof(float) * (size_t)BB * OC);
  float* fbuf = (float*)alloc(sizeof(float) * (size_t)BB * NP);
  float* mask = (float*)alloc(sizeof(float) * (size_t)BB * NP);

  const dim3 gtile(NP / 16, BB);

  // x0 = erase(x, ones, ones);  x1seq time-shift
  k_erase0<<<BB, 64, 0, stream>>>(x, w_erase, b_erase, g_erase, be_erase,
                                  m_erase, v_erase, x0);
  k_x1seq<<<(int)((feat_f + 255) / 256), 256, 0, stream>>>(x, x1s);

  // branch 0: kNN(x0,10) -> edge-conv -> p0, x0_vec
  k_xx<<<(BB * NP + 255) / 256, 256, 0, stream>>>(x0, xxb);
  k_knn<<<gtile, 32, 0, stream>>>(x0, xxb, idxm, KM);
  k_branch<<<gtile, 128, 0, stream>>>(x0, idxm, w0, g0, b0, m0, v0, ybuf);
  k_reduce_y<<<BB, 64, 0, stream>>>(ybuf, out, xvec, 0);

  // corre_binar: kNN(x2,8), f, argmax, masks
  k_xx<<<(BB * NP + 255) / 256, 256, 0, stream>>>(x1s, xxb);
  k_knn<<<gtile, 32, 0, stream>>>(x1s, xxb, idx8, KC);
  k_corre<<<BB, 64, 0, stream>>>(xvec, x1s, w_reduce, idx8, fbuf, mask);

  // branch 1: erase(x1s, masks, f) -> kNN(x1e,10) -> edge-conv -> p0+p1
  k_erase1<<<BB, 64, 0, stream>>>(x1s, mask, fbuf, w_erase, b_erase, g_erase,
                                  be_erase, m_erase, v_erase, x1e);
  k_xx<<<(BB * NP + 255) / 256, 256, 0, stream>>>(x1e, xxb);
  k_knn<<<gtile, 32, 0, stream>>>(x1e, xxb, idxm, KM);
  k_branch<<<gtile, 128, 0, stream>>>(x1e, idxm, w1, g1, b1, m1, v1, ybuf);
  k_reduce_y<<<BB, 64, 0, stream>>>(ybuf, out, nullptr, 1);
}
